// HMM_10866267259264
// MI455X (gfx1250) — matrix operand order repaired and checked by reading, not compiled, not run on hardware
//
#include <hip/hip_runtime.h>

// ---------------------------------------------------------------------------
// HMM forward scan on MI455X (gfx1250).
//   alpha_t = (alpha_{t-1} @ A) * em_t,  S=2048 states, T=8192 steps.
//
//  * A converted once to bf16 (bf16 keeps f32's exponent range -> matches the
//    geometric decay of unnormalized alphas) and prepacked into the exact
//    VGPR layout of the WMMA B-operand (32x16 bf16 tiles, lane-major so each
//    lane's 8 operand dwords are contiguous -> 2x ds_load_b128).
//  * 32 persistent workgroups x 128 threads (4 waves). Each WG owns 64
//    output columns; its 256 KB bf16 slice of A lives in LDS for the whole
//    kernel (320 KB LDS/WGP), fetched by ONE Tensor-Data-Mover op. A is read
//    from HBM exactly once.
//  * Per step, each wave computes 16 columns with a chain of 64
//    v_wmma_f32_16x16x32_bf16 ops (alpha broadcast in the A-operand,
//    A-matrix tile in the B-operand, f32 accumulation), as 4 independent
//    accumulator chains with a DEPTH-4 operand pipeline: each chain owns a
//    dedicated operand buffer refilled right after its WMMA issues, so every
//    operand fetch has 4 XDL slots of latency cover and WMMAs wait only on
//    dscnt<=12 (newest prefetches), never dscnt==0.
//  * Per-step 4 KB alpha broadcast is also a TDM op (tensor_load_to_lds of
//    the staging buffer into LDS, s_wait_tensorcnt + workgroup barrier).
//  * Cross-WG sync each step: monotonic atomic arrive counter in d_ws,
//    s_sleep spin, device-scope acquire/release.
//
// Workspace layout (needs ~8.02 MB):
//   [0, 8MB)          : packed bf16 A tiles, dword[64][128][32][8]
//   [8MB, 8MB+8KB)    : alpha staging, bf16[2][2048]
//   [8MB+8KB, +256B)  : barrier counter
// ---------------------------------------------------------------------------

#define S 2048
#define T 8192
#define NSYM 512
#define NBLK 32          // persistent workgroups
#define TPB 128          // 4 waves of 32
#define IBLK 64          // 2048 / 32 k-blocks
#define JT 128           // 2048 / 16 column tiles
#define PACKED_DWORDS (IBLK * JT * 32 * 8)   // 2,097,152 dwords = 8 MB

typedef __attribute__((ext_vector_type(16))) __bf16   v16bf;
typedef __attribute__((ext_vector_type(8)))  float    v8f;
typedef __attribute__((ext_vector_type(8)))  unsigned v8u;
typedef __attribute__((ext_vector_type(4)))  unsigned su4;

__device__ __forceinline__ unsigned bf16rne(float f) {
    unsigned u = __builtin_bit_cast(unsigned, f);
    return (u + 0x7fffu + ((u >> 16) & 1u)) >> 16;   // round-to-nearest-even
}

// ---------------------------------------------------------------------------
// Kernel 1: pack A (f32 row-major [2048][2048]) into bf16 WMMA B-operand
// tiles, lane-major. For tile (ib, jt), lane L, VGPR r (contiguous k order,
// same as the A-operand):
//   L < 16 : n = L,      k = {2r, 2r+1}
//   L >= 16: n = L - 16, k = {16+2r, 17+2r}
// dword = { lo: A[ib*32+k0][jt*16+n], hi: A[ib*32+k0+1][jt*16+n] }
// Linear index: d = ((ib*128 + jt)*32 + lane)*8 + r
// ---------------------------------------------------------------------------
__global__ void hmm_prepack(const float* __restrict__ A,
                            unsigned* __restrict__ packed) {
    for (int d = blockIdx.x * blockDim.x + threadIdx.x; d < PACKED_DWORDS;
         d += gridDim.x * blockDim.x) {
        const int r    = d & 7;
        const int lane = (d >> 3) & 31;
        const int jt   = (d >> 8) & 127;
        const int ib   = d >> 15;
        const int k0   = (lane < 16) ? (2 * r) : (16 + 2 * r);
        const int col  = jt * 16 + (lane & 15);
        const int row  = ib * 32 + k0;
        const unsigned lo = bf16rne(A[row * S + col]);
        const unsigned hi = bf16rne(A[(row + 1) * S + col]);
        packed[d] = lo | (hi << 16);
    }
}

// ---------------------------------------------------------------------------
// Monotonic grid barrier: each of the NBLK blocks arrives once per call;
// everyone waits until the global arrive count reaches `target`.
// ---------------------------------------------------------------------------
__device__ __forceinline__ void grid_sync(unsigned* bar, unsigned target) {
    __threadfence();                 // release our global writes (agent scope)
    __syncthreads();
    if (threadIdx.x == 0) {
        __hip_atomic_fetch_add(bar, 1u, __ATOMIC_ACQ_REL,
                               __HIP_MEMORY_SCOPE_AGENT);
        while (__hip_atomic_load(bar, __ATOMIC_ACQUIRE,
                                 __HIP_MEMORY_SCOPE_AGENT) < target)
            __builtin_amdgcn_s_sleep(1);
    }
    __syncthreads();
    __threadfence();                 // acquire others' writes
}

// Build TDM descriptor group 0 for a load into LDS at ldsByteAddr from ga.
__device__ __forceinline__ su4 tdm_g0(unsigned ldsByteAddr,
                                      unsigned long long ga) {
    su4 g0;
    g0[0] = 1u;                                   // count=1, no gather
    g0[1] = ldsByteAddr;                          // lds_addr
    g0[2] = (unsigned)ga;                         // global_addr[31:0]
    g0[3] = ((unsigned)(ga >> 32) & 0x01ffffffu)  // global_addr[56:32]
            | (2u << 30);                         // type=2 ("image")
    return g0;
}

// Build TDM descriptor group 1: 4-byte elements, tile tile0 x tile1,
// global row stride strid0 elements, huge tensor dims (no OOB clipping).
__device__ __forceinline__ v8u tdm_g1(unsigned tile0, unsigned tile1,
                                      unsigned strid0) {
    v8u g1;
    g1[0] = 2u << 16;               // data_size=4B; no multicast/pad/iterate
    g1[1] = 0u;                     // tensor_dim0[15:0]   (TD0 = 1<<30)
    g1[2] = 0x4000u;                // TD0[31:16], tensor_dim1[15:0]=0
    g1[3] = (tile0 << 16) | 0x10u;  // tile_dim0, TD1[31:16]=0x10 (TD1=1<<20)
    g1[4] = tile1;                  // tile_dim1, tile_dim2=0
    g1[5] = strid0;                 // tensor_dim0_stride[31:0]
    g1[6] = 0u;                     // stride0[47:32]=0, stride1[15:0]=0
    g1[7] = 0u;
    return g1;
}

// ---------------------------------------------------------------------------
// Kernel 2: persistent scan.
// ---------------------------------------------------------------------------
__global__ __launch_bounds__(TPB, 1)
void hmm_scan(const int* __restrict__ obs,
              const float* __restrict__ Bem,     // [2048][512]
              const float* __restrict__ pi,      // [2048]
              const unsigned* __restrict__ packedA,
              float* __restrict__ out,           // [8192][2048]
              unsigned short* __restrict__ staging,  // bf16[2][2048]
              unsigned* __restrict__ bar) {
    extern __shared__ __align__(16) unsigned smem[];
    unsigned* alphaLDS = smem;            // 1024 dwords = bf16[2048]
    unsigned* Bslice   = smem + 1024;     // 65536 dwords = 256 KB A slice

    const int tid  = threadIdx.x;
    const int lane = tid & 31;
    const int w    = tid >> 5;                 // wave 0..3
    const int jt0  = blockIdx.x * 4;           // this WG's first column tile
    const int j0   = blockIdx.x * 64 + w * 16; // this wave's first column
    const int jem  = j0 + (lane & 15);         // this lane's output column
    const unsigned ldsBase = __builtin_amdgcn_groupstaticsize();

    // ---- one-time TDM fetch of this WG's 256 KB bf16 A slice into LDS ----
    // 2-D tile: 64 rows (ib) x 1024 dwords, global row stride 32768 dwords,
    // packed contiguously into LDS at Bslice. Issued by wave 0 only.
    if (tid < 32) {
        const su4 g0 = tdm_g0(
            ldsBase + 4096u,
            (unsigned long long)(const void*)(packedA + (size_t)jt0 * 256));
        const v8u g1 = tdm_g1(1024u, 64u, 32768u);
        asm volatile("tensor_load_to_lds %0, %1"
                     :: "s"(g0), "s"(g1) : "memory");
        __builtin_amdgcn_s_wait_tensorcnt(0);
    }

    // ---- alpha0 = pi * em0 ----
    {
        const int o0 = __builtin_amdgcn_readfirstlane(obs[0]);
        if (tid < 64) {
            const int j = blockIdx.x * 64 + tid;
            const float v = pi[j] * Bem[j * NSYM + o0];
            out[j] = v;
            staging[0 * S + j] = (unsigned short)bf16rne(v);
        }
    }
    __syncthreads();                 // LDS slice visible to all waves
    unsigned arrivals = 0;
    grid_sync(bar, ++arrivals * NBLK);

    // per-step alpha broadcast descriptors (double-buffered staging)
    const su4 g0Alpha0 =
        tdm_g0(ldsBase, (unsigned long long)(const void*)staging);
    const su4 g0Alpha1 =
        tdm_g0(ldsBase, (unsigned long long)(const void*)(staging + S));
    const v8u g1Alpha = tdm_g1(1024u, 1u, 1024u);

    const unsigned  abash  = (lane >= 16) ? 8u : 0u;   // K-half select (A-op)
    const unsigned* btbase = Bslice + (unsigned)w * 256 + (unsigned)lane * 8;

    // operand fetch for k-block ib: A-op = alpha broadcast (VGPR r holds
    // bf16 pair k={kbase,kbase+1}, kbase = 2r + 16*half);
    // B-op = prepacked lane-major 32x16 bf16 tile.
    auto loadops = [&](int ib, v8u& au, v8u& bu) {
        const unsigned abase = (unsigned)(ib * 16) + abash;
        const uint4 x0 = *(const uint4*)(alphaLDS + abase);
        const uint4 x1 = *(const uint4*)(alphaLDS + abase + 4);
        v8u a = {x0.x, x0.y, x0.z, x0.w, x1.x, x1.y, x1.z, x1.w};
        const unsigned* bt = btbase + (unsigned)(ib * 1024);
        const uint4 y0 = *(const uint4*)bt;
        const uint4 y1 = *(const uint4*)(bt + 4);
        v8u b = {y0.x, y0.y, y0.z, y0.w, y1.x, y1.y, y1.z, y1.w};
        au = a; bu = b;
    };

    for (int t = 1; t < T; ++t) {
        // broadcast alpha_{t-1} (bf16, 4 KB) into LDS via TDM (wave 0)
        if (tid < 32) {
            if (((t - 1) & 1) == 0)
                asm volatile("tensor_load_to_lds %0, %1"
                             :: "s"(g0Alpha0), "s"(g1Alpha) : "memory");
            else
                asm volatile("tensor_load_to_lds %0, %1"
                             :: "s"(g0Alpha1), "s"(g1Alpha) : "memory");
            __builtin_amdgcn_s_wait_tensorcnt(0);
        }
        __syncthreads();

        const int   ot = __builtin_amdgcn_readfirstlane(obs[t]);
        const float em = Bem[jem * NSYM + ot];

        // 64-block K-chain: 4 WMMA accumulator chains, depth-4 pipeline.
        // Chain ch owns operand buffer (a[ch], b[ch]); it is refilled for
        // the next ib2 right after chain ch's WMMA issues, giving each DS
        // fetch 4 WMMA slots of latency cover (wait dscnt<=12, never 0).
        v8f c[4] = {v8f{}, v8f{}, v8f{}, v8f{}};
        v8u a[4], b[4];
#pragma unroll
        for (int ch = 0; ch < 4; ++ch)
            loadops(ch * 16, a[ch], b[ch]);
#pragma unroll 1
        for (int ib2 = 0; ib2 < 16; ++ib2) {
#pragma unroll
            for (int ch = 0; ch < 4; ++ch) {
                c[ch] = __builtin_amdgcn_wmma_f32_16x16x32_bf16(
                    false, __builtin_bit_cast(v16bf, a[ch]),
                    false, __builtin_bit_cast(v16bf, b[ch]),
                    (short)0, c[ch], false, false);
                // refill this chain's buffer for the next k-block
                // (wraps to ib2=0 at the end: dead loads, branch-free loop)
                const int nib = ch * 16 + ((ib2 + 1) & 15);
                loadops(nib, a[ch], b[ch]);
            }
        }

        // every lane's D-VGPR0 holds out[j0 + (lane&15)] (all M rows equal)
        const float dot = (c[0][0] + c[1][0]) + (c[2][0] + c[3][0]);
        const float val = dot * em;
        if (lane < 16) {
            out[t * S + j0 + lane] = val;
            staging[((unsigned)t & 1u) * S + j0 + lane] =
                (unsigned short)bf16rne(val);
        }
        grid_sync(bar, ++arrivals * NBLK);
    }
}

// ---------------------------------------------------------------------------
extern "C" void kernel_launch(void* const* d_in, const int* in_sizes, int n_in,
                              void* d_out, int out_size, void* d_ws,
                              size_t ws_size, hipStream_t stream) {
    (void)in_sizes; (void)n_in; (void)out_size; (void)ws_size;
    const int*   obs = (const int*)d_in[0];
    const float* A   = (const float*)d_in[1];
    const float* Bem = (const float*)d_in[2];
    const float* pi  = (const float*)d_in[3];
    float*       out = (float*)d_out;

    char* ws = (char*)d_ws;
    unsigned*       packedA = (unsigned*)ws;                       // 8 MB
    unsigned short* staging = (unsigned short*)(ws + (8u << 20));  // 8 KB
    unsigned*       bar     = (unsigned*)(ws + (8u << 20) + 8192); // 256 B

    hipMemsetAsync(bar, 0, 256, stream);                 // reset barrier
    hmm_prepack<<<2048, 256, 0, stream>>>(A, packedA);   // A -> bf16 tiles

    const size_t lds_bytes = 4096 + 262144;              // alpha + A slice
    hmm_scan<<<NBLK, TPB, lds_bytes, stream>>>(obs, Bem, pi, packedA, out,
                                               staging, bar);
}
